// StylizationBlock_84318797955700
// MI455X (gfx1250) — matrix operand order repaired and checked by reading, not compiled, not run on hardware
//
#include <hip/hip_runtime.h>
#include <hip/hip_bf16.h>

// B=16, P=16, D=128, T=128
#define NEGV -1e9f

typedef __attribute__((ext_vector_type(16))) _Float16 v16h;
typedef __attribute__((ext_vector_type(8)))  _Float16 v8h;
typedef __attribute__((ext_vector_type(4)))  _Float16 v4h;
typedef __attribute__((ext_vector_type(8)))  float    v8f;
typedef __attribute__((ext_vector_type(4)))  float    v4f;

// ---- WMMA fragment helpers (CDNA5 16x16x32 f16 layouts) ----
// A (16x32 f16): lane<16 row=lane, elems 0..7 = K k0..k0+7, elems 8..15 = K k0+16..k0+23
//                lane>=16 same row, K shifted by +8 within each group (hi=1)
__device__ __forceinline__ v16h frag_a_f16(const _Float16* A, int lda, int row, int k0, int hi) {
  const _Float16* p = A + (size_t)row * lda + k0 + hi * 8;
  v8h lo = *(const v8h*)p;
  v8h up = *(const v8h*)(p + 16);
  v16h a;
#pragma unroll
  for (int i = 0; i < 8; ++i) { a[i] = lo[i]; a[i + 8] = up[i]; }
  return a;
}
// B (32x16 f16), weights stored transposed [N][K]: lane<16 col=lane, elems 0..7 = K k0..k0+7, elems 8..15 = K k0+16..k0+23;
// lane>=16 col=lane-16, K shifted by +8 within each group
__device__ __forceinline__ v16h frag_b_f16(const _Float16* Wt, int ldk, int n, int k0, int hi) {
  const _Float16* p = Wt + (size_t)n * ldk + k0 + hi * 8;
  v8h b0 = *(const v8h*)p;
  v8h b1 = *(const v8h*)(p + 16);
  v16h b;
#pragma unroll
  for (int i = 0; i < 8; ++i) { b[i] = b0[i]; b[i + 8] = b1[i]; }
  return b;
}
__device__ __forceinline__ v8f wmma_f16(v16h a, v16h b, v8f c) {
  return __builtin_amdgcn_wmma_f32_16x16x32_f16(false, a, false, b, (short)0, c, false, false);
}

// ---- prep: transpose+convert f32 W[K][N] -> f16 Wt[N][K] ----
__global__ void tr_convert_kernel(const float* __restrict__ W, _Float16* __restrict__ Wt,
                                  int K, int N) {
  int idx = blockIdx.x * 256 + threadIdx.x;
  if (idx >= K * N) return;
  int n = idx / K, k = idx - n * K;
  Wt[idx] = (_Float16)W[(size_t)k * N + n];
}

// ---- prep: u[e] = sum_d w2[e][d]*dw[d]; u[128] = c = b2.dw + db ----
__global__ void prep_u_kernel(const float* __restrict__ w2, const float* __restrict__ dw,
                              const float* __restrict__ b2, const float* __restrict__ db,
                              float* __restrict__ u) {
  int e = threadIdx.x;  // 128
  float acc = 0.f;
  for (int d = 0; d < 128; ++d) acc += w2[(size_t)e * 128 + d] * dw[d];
  u[e] = acc;
  if (e == 0) {
    float cc = 0.f;
    for (int d = 0; d < 128; ++d) cc += b2[d] * dw[d];
    u[128] = cc + db[0];
  }
}

// ---- person pooling scores: scores[b,p] = mean_t(xg[b,p,:,:]).pool_w + pool_b, masked ----
__global__ void attn_scores_kernel(const float* __restrict__ xg, const float* __restrict__ pool_w,
                                   const float* __restrict__ pool_b, const int* __restrict__ pmask,
                                   float* __restrict__ scores) {
  int bp = blockIdx.x;  // 256
  const float* base = xg + (size_t)bp * 16384;
  float acc = 0.f;
  for (int i = threadIdx.x; i < 16384; i += 256) acc += base[i] * pool_w[i >> 7];
  __shared__ float red[256];
  red[threadIdx.x] = acc;
  __syncthreads();
  for (int s = 128; s > 0; s >>= 1) {
    if (threadIdx.x < s) red[threadIdx.x] += red[threadIdx.x + s];
    __syncthreads();
  }
  if (threadIdx.x == 0) {
    float sc = red[0] * (1.f / 128.f) + pool_b[0];
    scores[bp] = (pmask[bp] == 0) ? NEGV : sc;
  }
}

__global__ void attn_softmax_kernel(const float* __restrict__ scores, float* __restrict__ attn) {
  int tid = threadIdx.x;  // 256 = B*P
  __shared__ float s[256];
  s[tid] = scores[tid];
  __syncthreads();
  int b = tid >> 4;
  float mx = -3.4e38f;
  for (int q = 0; q < 16; ++q) mx = fmaxf(mx, s[b * 16 + q]);
  float den = 0.f;
  for (int q = 0; q < 16; ++q) den += __expf(s[b * 16 + q] - mx);
  attn[tid] = __expf(s[tid] - mx) / den;
}

// ---- xg_pooled[b,d,t] = sum_p attn[b,p]*xg[b,p,d,t]  (f16 rows (b*128+d) x t) ----
__global__ void pool_xg_kernel(const float* __restrict__ xg, const float* __restrict__ attn,
                               _Float16* __restrict__ pooled) {
  int idx = blockIdx.x * 256 + threadIdx.x;  // 262144
  int t = idx & 127, d = (idx >> 7) & 127, b = idx >> 14;
  const float* base = xg + ((size_t)(b * 16) * 128 + d) * 128 + t;
  float acc = 0.f;
  for (int p = 0; p < 16; ++p) acc += attn[b * 16 + p] * base[(size_t)p * 16384];
  pooled[idx] = (_Float16)acc;
}

// ---- emb GEMM: (2048 x 128) @ (128 x 256) + emb_b -> embout f32 (b,d,0..255)
//      N-register-blocked: one A load serves all 16 N-tiles ----
__global__ void __launch_bounds__(256)
gemm_emb_kernel(const _Float16* __restrict__ A, const _Float16* __restrict__ Wt,
                const float* __restrict__ bias, float* __restrict__ C) {
  const int wave = threadIdx.x >> 5, lane = threadIdx.x & 31;
  const int tm = blockIdx.x * 8 + wave;  // 128 M-tiles
  const int lr = lane & 15, hi = (lane >> 4) & 1;
  const int row = tm * 16 + lr;
  v16h af[4];
#pragma unroll
  for (int c = 0; c < 4; ++c) af[c] = frag_a_f16(A, 128, row, c * 32, hi);
  for (int tn = 0; tn < 16; ++tn) {
    const int ncol = tn * 16 + lr;
    v8f acc = {};
#pragma unroll
    for (int c = 0; c < 4; ++c)
      acc = wmma_f16(af[c], frag_b_f16(Wt, 128, ncol, c * 32, hi), acc);
    const float bn = bias[ncol];
#pragma unroll
    for (int v = 0; v < 8; ++v) {
      int m = tm * 16 + hi * 8 + v;
      C[(size_t)m * 256 + ncol] = acc[v] + bn;
    }
  }
}

// ---- fused out/glob GEMMs over T: rows (b,p,d)=32768, K=T=128, N=128
//      one x/scale/shift load serves all 8 N-tiles for both GEMMs ----
__global__ void __launch_bounds__(256)
gemm_out_glob_kernel(const float* __restrict__ x, const float* __restrict__ xg,
                     const float* __restrict__ embout,
                     const _Float16* __restrict__ outWt, const _Float16* __restrict__ globWt,
                     const float* __restrict__ out_b, const float* __restrict__ glob_b,
                     float* __restrict__ x1, float* __restrict__ xg2) {
  const int wave = threadIdx.x >> 5, lane = threadIdx.x & 31;
  const int tm = blockIdx.x * 8 + wave;  // 2048 M-tiles
  const int lr = lane & 15, hi = (lane >> 4) & 1;
  const int row = tm * 16 + lr;                 // (b,p,d)
  const int b = row >> 11, d = row & 127;
  const size_t srow = (size_t)(b * 128 + d) * 256;
  v16h ax[4], am[4];
#pragma unroll
  for (int c = 0; c < 4; ++c) {
    const int kk = c * 32 + hi * 8;
    const float* px = x + (size_t)row * 128 + kk;
    v4f x0 = *(const v4f*)px,        x1v = *(const v4f*)(px + 4);
    v4f x2 = *(const v4f*)(px + 16), x3  = *(const v4f*)(px + 20);
    const float* psc = embout + srow + kk;      // scale = embout[...,k]
    v4f s0 = *(const v4f*)psc,        s1 = *(const v4f*)(psc + 4);
    v4f s2 = *(const v4f*)(psc + 16), s3 = *(const v4f*)(psc + 20);
    const float* psh = psc + 128;               // shift = embout[...,128+k]
    v4f h0 = *(const v4f*)psh,        h1 = *(const v4f*)(psh + 4);
    v4f h2 = *(const v4f*)(psh + 16), h3 = *(const v4f*)(psh + 20);
#pragma unroll
    for (int i = 0; i < 4; ++i) {
      ax[c][i]      = (_Float16)x0[i];
      ax[c][4 + i]  = (_Float16)x1v[i];
      ax[c][8 + i]  = (_Float16)x2[i];
      ax[c][12 + i] = (_Float16)x3[i];
      am[c][i]      = (_Float16)(x0[i]  * (1.f + s0[i]) + h0[i]);
      am[c][4 + i]  = (_Float16)(x1v[i] * (1.f + s1[i]) + h1[i]);
      am[c][8 + i]  = (_Float16)(x2[i]  * (1.f + s2[i]) + h2[i]);
      am[c][12 + i] = (_Float16)(x3[i]  * (1.f + s3[i]) + h3[i]);
    }
  }
  for (int tn = 0; tn < 8; ++tn) {
    const int ncol = tn * 16 + lr;
    v8f accO = {}, accG = {};
#pragma unroll
    for (int c = 0; c < 4; ++c) {
      accO = wmma_f16(am[c], frag_b_f16(outWt, 128, ncol, c * 32, hi), accO);
      accG = wmma_f16(ax[c], frag_b_f16(globWt, 128, ncol, c * 32, hi), accG);
    }
    const float ob = out_b[ncol], gb = glob_b[ncol];
#pragma unroll
    for (int v = 0; v < 8; ++v) {
      int m = tm * 16 + hi * 8 + v;
      size_t idx = (size_t)m * 128 + ncol;
      x1[idx]  = accO[v] + ob;
      xg2[idx] = xg[idx] + accG[v] + gb;
    }
  }
}

// ---- distance attention (algebraically reduced): per (b,p,t) compute softmax over q
//      of relu(s_q*w1+b1).u + c and pool hbar[d] = sum_q aw_q*relu(s_q*w1[d]+b1[d]) ----
__global__ void __launch_bounds__(128)
dist_attn_kernel(const float* __restrict__ dist, const int* __restrict__ pmask,
                 const float* __restrict__ w1, const float* __restrict__ b1,
                 const float* __restrict__ u, _Float16* __restrict__ hbar) {
  const int bp = blockIdx.x, b = bp >> 4, p = bp & 15;
  const int t = threadIdx.x;  // 128
  __shared__ float sw1[128], sb1[128], su[128], smask[16];
  sw1[t] = w1[t]; sb1[t] = b1[t]; su[t] = u[t];
  if (t < 16) smask[t] = (pmask[b * 16 + t] == 0) ? 0.f : 1.f;
  __syncthreads();
  const float c = u[128];
  float s[16], lg[16];
  const float* dp = dist + (((size_t)b * 128 + t) * 16 + p) * 16;  // distances[b][t][p][q]
#pragma unroll
  for (int q = 0; q < 16; ++q) s[q] = dp[q];
  for (int q = 0; q < 16; ++q) {
    float acc = 0.f;
    for (int d = 0; d < 128; ++d) acc += fmaxf(s[q] * sw1[d] + sb1[d], 0.f) * su[d];
    lg[q] = (smask[q] != 0.f) ? (acc + c) : NEGV;
  }
  float mx = -3.4e38f;
#pragma unroll
  for (int q = 0; q < 16; ++q) mx = fmaxf(mx, lg[q]);
  float e[16], den = 0.f;
#pragma unroll
  for (int q = 0; q < 16; ++q) { e[q] = __expf(lg[q] - mx); den += e[q]; }
  const float inv = 1.f / den;
  _Float16* out = hbar + ((size_t)bp * 128 + t) * 128;
  for (int d = 0; d < 128; d += 4) {
    v4h pk;
#pragma unroll
    for (int j = 0; j < 4; ++j) {
      float w = sw1[d + j], bb = sb1[d + j], hb = 0.f;
#pragma unroll
      for (int q = 0; q < 16; ++q) hb += e[q] * fmaxf(s[q] * w + bb, 0.f);
      pk[j] = (_Float16)(hb * inv);
    }
    *(v4h*)(out + d) = pk;
  }
}

// ---- d_ctx GEMM: (32768 rows (bp,t) x 128) @ w2 (128x128) + b2, transposed f16 epilogue ----
__global__ void __launch_bounds__(256)
gemm_dctx_kernel(const _Float16* __restrict__ hbar, const _Float16* __restrict__ w2T,
                 const float* __restrict__ b2, _Float16* __restrict__ dctxT) {
  const int wave = threadIdx.x >> 5, lane = threadIdx.x & 31;
  const int tm = blockIdx.x * 8 + wave;  // 2048 M-tiles
  const int lr = lane & 15, hi = (lane >> 4) & 1;
  const int row = tm * 16 + lr;
  v16h af[4];
#pragma unroll
  for (int c = 0; c < 4; ++c) af[c] = frag_a_f16(hbar, 128, row, c * 32, hi);
  const int t0 = (tm * 16 + hi * 8) & 127;   // 8 consecutive t values per lane
  const int bpi = (tm * 16) >> 7;            // same bp for whole tile (128 | 16-tiles)
  for (int tn = 0; tn < 8; ++tn) {
    const int ncol = tn * 16 + lr;
    v8f acc = {};
#pragma unroll
    for (int c = 0; c < 4; ++c)
      acc = wmma_f16(af[c], frag_b_f16(w2T, 128, ncol, c * 32, hi), acc);
    const float bn = b2[ncol];
    v8h pk;
#pragma unroll
    for (int v = 0; v < 8; ++v) pk[v] = (_Float16)(acc[v] + bn);
    *(v8h*)(dctxT + ((size_t)(bpi * 128 + ncol)) * 128 + t0) = pk;  // [b,p,d,t]
  }
}

// ---- temp GEMM over T + residual add into x1 (in place) ----
__global__ void __launch_bounds__(256)
gemm_temp_kernel(const _Float16* __restrict__ dctxT, const _Float16* __restrict__ tempWt,
                 const float* __restrict__ temp_b, float* __restrict__ x1) {
  const int wave = threadIdx.x >> 5, lane = threadIdx.x & 31;
  const int tm = blockIdx.x * 8 + wave;  // 2048 M-tiles
  const int lr = lane & 15, hi = (lane >> 4) & 1;
  const int row = tm * 16 + lr;
  v16h af[4];
#pragma unroll
  for (int c = 0; c < 4; ++c) af[c] = frag_a_f16(dctxT, 128, row, c * 32, hi);
  for (int tn = 0; tn < 8; ++tn) {
    const int ncol = tn * 16 + lr;
    v8f acc = {};
#pragma unroll
    for (int c = 0; c < 4; ++c)
      acc = wmma_f16(af[c], frag_b_f16(tempWt, 128, ncol, c * 32, hi), acc);
    const float bn = temp_b[ncol];
#pragma unroll
    for (int v = 0; v < 8; ++v) {
      int m = tm * 16 + hi * 8 + v;
      size_t idx = (size_t)m * 128 + ncol;
      x1[idx] = x1[idx] + acc[v] + bn;
    }
  }
}

// ---- layernorm over D axis, in place: block=(b,p), thread=t ----
__global__ void __launch_bounds__(128)
ln_kernel(float* __restrict__ buf, const float* __restrict__ alpha,
          const float* __restrict__ beta) {
  const int bp = blockIdx.x, t = threadIdx.x;
  float* base = buf + (size_t)bp * 16384 + t;
  float sum = 0.f, sq = 0.f;
  for (int d = 0; d < 128; ++d) {
    float v = base[(size_t)d * 128];
    sum += v; sq += v * v;
  }
  const float mean = sum * (1.f / 128.f);
  const float var = sq * (1.f / 128.f) - mean * mean;
  const float rstd = rsqrtf(var + 1e-5f);
  for (int d = 0; d < 128; ++d) {
    float v = base[(size_t)d * 128];
    base[(size_t)d * 128] = (v - mean) * rstd * alpha[d] + beta[d];
  }
}

extern "C" void kernel_launch(void* const* d_in, const int* in_sizes, int n_in,
                              void* d_out, int out_size, void* d_ws, size_t ws_size,
                              hipStream_t stream) {
  (void)in_sizes; (void)n_in; (void)out_size; (void)ws_size;
  const float* x        = (const float*)d_in[0];
  const float* xg       = (const float*)d_in[1];
  const float* dist     = (const float*)d_in[2];
  const int*   pmask    = (const int*)d_in[3];
  const float* emb_w    = (const float*)d_in[4];
  const float* emb_b    = (const float*)d_in[5];
  const float* out_w    = (const float*)d_in[6];
  const float* out_b    = (const float*)d_in[7];
  const float* glob_w   = (const float*)d_in[8];
  const float* glob_b   = (const float*)d_in[9];
  const float* temp_w   = (const float*)d_in[10];
  const float* temp_b   = (const float*)d_in[11];
  const float* mlp_w1   = (const float*)d_in[12];
  const float* mlp_b1   = (const float*)d_in[13];
  const float* mlp_w2   = (const float*)d_in[14];
  const float* mlp_b2   = (const float*)d_in[15];
  const float* pool_w   = (const float*)d_in[16];
  const float* pool_b   = (const float*)d_in[17];
  const float* dis_w    = (const float*)d_in[18];
  const float* dis_b    = (const float*)d_in[19];
  const float* n_alpha  = (const float*)d_in[20];
  const float* n_beta   = (const float*)d_in[21];
  const float* g_alpha  = (const float*)d_in[22];
  const float* g_beta   = (const float*)d_in[23];

  char* wp = (char*)d_ws;
  auto take = [&](size_t bytes) { char* r = wp; wp += (bytes + 255) & ~(size_t)255; return r; };
  float*     scores = (float*)take(256 * 4);
  float*     attn   = (float*)take(256 * 4);
  float*     embout = (float*)take((size_t)16 * 128 * 256 * 4);   // (b,d,2T)
  _Float16*  pooled = (_Float16*)take((size_t)2048 * 128 * 2);
  _Float16*  outWt  = (_Float16*)take(16384 * 2);
  _Float16*  globWt = (_Float16*)take(16384 * 2);
  _Float16*  tempWt = (_Float16*)take(16384 * 2);
  _Float16*  w2T    = (_Float16*)take(16384 * 2);
  _Float16*  embWt  = (_Float16*)take(32768 * 2);
  float*     u      = (float*)take(129 * 4);
  _Float16*  hbar   = (_Float16*)take((size_t)32768 * 128 * 2);
  _Float16*  dctxT  = (_Float16*)take((size_t)32768 * 128 * 2);

  float* x1  = (float*)d_out;                 // (B,P,D,T) output half 1, used as scratch
  float* xg2 = x1 + (size_t)16 * 16 * 128 * 128;  // output half 2

  tr_convert_kernel<<<64, 256, 0, stream>>>(out_w, outWt, 128, 128);
  tr_convert_kernel<<<64, 256, 0, stream>>>(glob_w, globWt, 128, 128);
  tr_convert_kernel<<<64, 256, 0, stream>>>(temp_w, tempWt, 128, 128);
  tr_convert_kernel<<<64, 256, 0, stream>>>(mlp_w2, w2T, 128, 128);
  tr_convert_kernel<<<128, 256, 0, stream>>>(emb_w, embWt, 128, 256);
  prep_u_kernel<<<1, 128, 0, stream>>>(mlp_w2, dis_w, mlp_b2, dis_b, u);

  attn_scores_kernel<<<256, 256, 0, stream>>>(xg, pool_w, pool_b, pmask, scores);
  attn_softmax_kernel<<<1, 256, 0, stream>>>(scores, attn);
  pool_xg_kernel<<<1024, 256, 0, stream>>>(xg, attn, pooled);
  gemm_emb_kernel<<<16, 256, 0, stream>>>(pooled, embWt, emb_b, embout);

  gemm_out_glob_kernel<<<256, 256, 0, stream>>>(x, xg, embout, outWt, globWt,
                                                out_b, glob_b, x1, xg2);

  dist_attn_kernel<<<256, 128, 0, stream>>>(dist, pmask, mlp_w1, mlp_b1, u, hbar);
  gemm_dctx_kernel<<<256, 256, 0, stream>>>(hbar, w2T, mlp_b2, dctxT);
  gemm_temp_kernel<<<256, 256, 0, stream>>>(dctxT, tempWt, temp_b, x1);

  ln_kernel<<<256, 128, 0, stream>>>(x1, n_alpha, n_beta);
  ln_kernel<<<256, 128, 0, stream>>>(xg2, g_alpha, g_beta);
}
